// GroupWiseCorrelationCostVolume_15822659518755
// MI455X (gfx1250) — matrix operand order repaired and checked by reading, not compiled, not run on hardware
//
#include <hip/hip_runtime.h>
#include <hip/hip_bf16.h>

// ---------------------------------------------------------------------------
// GroupWiseCorrelationCostVolume + 4x (Conv3d 3^3 + BN(train) + LeakyReLU) + Conv3d->1
// MI455X (gfx1250, wave32). Conv layers run on v_wmma_f32_16x16x32_bf16.
// Activations live channels-last bf16; weights pre-packed to WMMA A-layout.
// Halo staging uses GLOBAL_LOAD_ASYNC_TO_LDS (ASYNCcnt) when available.
// ---------------------------------------------------------------------------

typedef __attribute__((ext_vector_type(16))) __bf16 v16bf;
typedef __attribute__((ext_vector_type(8)))  float  v8f;
typedef int gv4i __attribute__((vector_size(16)));   // matches builtin param pointee

#define BB   2
#define CC   128
#define HH   64
#define WW   128
#define GG   16
#define CPG  8
#define DD   48
#define HID  32
#define NPOS (BB*DD*HH*WW)        // 786432 output positions per layer
#define PACKSZ (2*27*32*16)       // u16 per packed layer: [mtile][t][lane][16]

#if defined(__AMDGCN__) && __has_builtin(__builtin_amdgcn_global_load_async_to_lds_b128) && __has_builtin(__builtin_amdgcn_s_wait_asynccnt)
#define USE_ASYNC 1
#else
#define USE_ASYNC 0
#endif

__device__ __forceinline__ float bf2f(unsigned short u) {
  unsigned int x = ((unsigned int)u) << 16;
  return __builtin_bit_cast(float, x);
}
__device__ __forceinline__ unsigned short f2bf(float f) {
  unsigned int x = __builtin_bit_cast(unsigned int, f);
  x += 0x7FFFu + ((x >> 16) & 1u);          // round-to-nearest-even
  return (unsigned short)(x >> 16);
}

#if USE_ASYNC
__device__ __forceinline__ void async_g2l_b128(const void* g, void* l) {
  __builtin_amdgcn_global_load_async_to_lds_b128(
      (__attribute__((address_space(1))) gv4i*)g,
      (__attribute__((address_space(3))) gv4i*)l, 0, 0);
}
#endif

// ---------------------------------------------------------------------------
// Zero the BN statistics scratch (4 layers x [32 sum | 32 sumsq]).
__global__ void k_init(float* __restrict__ st) {
  st[threadIdx.x] = 0.f;
}

// ---------------------------------------------------------------------------
// Pack conv weights [O][I][3][3][3] f32 -> bf16 WMMA A-matrix lane layout.
// Dest u16 index (per layer): ((mtile*27 + t)*32 + lane)*16 + j
// A 16x32 layout (contiguous K): lane<16 holds K 0..15, lane>=16 holds K 16..31.
// Layer 0 has Cin=16; channels 16..31 padded with zero weights.
__global__ void k_pack(const float* __restrict__ w1, const float* __restrict__ w2,
                       const float* __restrict__ w3, const float* __restrict__ w4,
                       unsigned short* __restrict__ wpack) {
  int id = blockIdx.x * 256 + threadIdx.x;
  if (id >= 4 * PACKSZ) return;
  int layer = id / PACKSZ;
  int r     = id % PACKSZ;
  int j     = r & 15;
  int lane  = (r >> 4) & 31;
  int tt    = r >> 9;             // 0..53
  int t     = tt % 27;
  int mt    = tt / 27;
  int m     = mt * 16 + (lane & 15);
  int k     = ((lane >> 4) << 4) + j;
  const float* w = (layer == 0) ? w1 : (layer == 1) ? w2 : (layer == 2) ? w3 : w4;
  int Cin = (layer == 0) ? 16 : 32;
  float v = (k < Cin) ? w[((size_t)m * Cin + k) * 27 + t] : 0.f;
  wpack[id] = f2bf(v);
}

// ---------------------------------------------------------------------------
// Group-wise correlation volume -> ACT bf16 channels-last [B,D,H,W,32]
// (16 real groups + 16 zero pad so conv layer 1 uses the uniform K=32 kernel).
// One block = (b, h, 64-wide W segment). LDS staged CHANNEL-CONTIGUOUS
// ([w][c], row padded to 136 u16 = 68 dwords to stagger banks) so each
// 8-channel group dot is one ds_load_b128 per operand instead of 16 u16 loads.
__global__ void k_volume(const float* __restrict__ L, const float* __restrict__ R,
                         unsigned short* __restrict__ act) {
  __shared__ __align__(16) unsigned short ls[64][136];    // w-local x channel
  __shared__ __align__(16) unsigned short rs[112][136];   // w' = w0-47 .. w0+64
  int bid = blockIdx.x;
  int seg = bid & 1;
  int rest = bid >> 1;
  int h = rest & (HH - 1);
  int b = rest >> 6;
  int tid = threadIdx.x;
  int w0 = seg * 64;

  // stage left: read [c][w] rows vectorized, scatter-transpose into [w][c]
  for (int ch = tid; ch < CC * 16; ch += 256) {
    int c = ch >> 4, j = ch & 15;
    const float4 f4 = *(const float4*)(L + (((size_t)b * CC + c) * HH + h) * WW + w0 + j * 4);
    ls[j*4+0][c] = f2bf(f4.x); ls[j*4+1][c] = f2bf(f4.y);
    ls[j*4+2][c] = f2bf(f4.z); ls[j*4+3][c] = f2bf(f4.w);
  }
  // stage right with disparity halo, zero-fill w'<0
  for (int ch = tid; ch < CC * 112; ch += 256) {
    int c = ch / 112, i = ch % 112;
    int wg = w0 - 47 + i;
    float v = (wg >= 0 && wg < WW) ? R[(((size_t)b * CC + c) * HH + h) * WW + wg] : 0.f;
    rs[i][c] = f2bf(v);
  }
  __syncthreads();

  for (int it = tid; it < DD * 64; it += 256) {
    int d = it >> 6, wl = it & 63;
    int ri = wl - d + 47;                       // index into rs (>=0 by construction)
    const uint4* lrow = (const uint4*)(&ls[wl][0]);   // 16 uint4 = 16 groups of 8ch
    const uint4* rrow = (const uint4*)(&rs[ri][0]);
    float acc[GG];
    #pragma unroll
    for (int g = 0; g < GG; ++g) {
      uint4 lu = lrow[g];
      uint4 ru = rrow[g];
      float s;
      s  = bf2f((unsigned short)lu.x) * bf2f((unsigned short)ru.x);
      s += bf2f((unsigned short)(lu.x >> 16)) * bf2f((unsigned short)(ru.x >> 16));
      s += bf2f((unsigned short)lu.y) * bf2f((unsigned short)ru.y);
      s += bf2f((unsigned short)(lu.y >> 16)) * bf2f((unsigned short)(ru.y >> 16));
      s += bf2f((unsigned short)lu.z) * bf2f((unsigned short)ru.z);
      s += bf2f((unsigned short)(lu.z >> 16)) * bf2f((unsigned short)(ru.z >> 16));
      s += bf2f((unsigned short)lu.w) * bf2f((unsigned short)ru.w);
      s += bf2f((unsigned short)(lu.w >> 16)) * bf2f((unsigned short)(ru.w >> 16));
      acc[g] = s * 0.125f;                      // mean over cpg=8
    }
    size_t pos = ((size_t)((b * DD + d) * HH + h)) * WW + (w0 + wl);
    unsigned int pk[8];
    #pragma unroll
    for (int g = 0; g < 8; ++g)
      pk[g] = (unsigned int)f2bf(acc[2*g]) | ((unsigned int)f2bf(acc[2*g+1]) << 16);
    uint4* dst = (uint4*)(act + pos * 32);
    dst[0] = make_uint4(pk[0], pk[1], pk[2], pk[3]);
    dst[1] = make_uint4(pk[4], pk[5], pk[6], pk[7]);
    dst[2] = make_uint4(0, 0, 0, 0);
    dst[3] = make_uint4(0, 0, 0, 0);
  }
}

// ---------------------------------------------------------------------------
// WMMA conv layer: ACT bf16 [*,32] -> Y f32 [*,32], plus per-channel sum/sumsq.
// Block = 128 threads (4 waves); tile = (b,d,h, 64-wide W segment), each wave
// owns 16 W positions. K loop = 27 kernel offsets x 32 channels; 2 M-tiles.
// Halo staging: async global->LDS (ASYNCcnt) for valid chunks, ds zero-fill OOB.
__global__ void __launch_bounds__(128) k_conv(const unsigned short* __restrict__ act,
                                              float* __restrict__ y,
                                              const unsigned short* __restrict__ wp,
                                              float* __restrict__ stats) {
  __shared__ __align__(32) unsigned short sm[9][66][32];   // (dz,dy) x (w halo) x ch
  __shared__ float sred[64];
  int tid = threadIdx.x;
  int bid = blockIdx.x;
  int seg = bid & 1;
  int rest = bid >> 1;
  int h = rest & 63;
  int d = (rest >> 6) % DD;
  int b = (rest >> 6) / DD;
  int w0 = seg * 64;
  if (tid < 64) sred[tid] = 0.f;

  // stage 9 halo rows x 66 positions x 32 bf16 (16B chunks), zero OOB
  for (int ch = tid; ch < 9 * 66 * 4; ch += 128) {
    int r   = ch / (66 * 4);
    int rem = ch - r * (66 * 4);
    int pi  = rem >> 2;
    int q   = rem & 3;
    int sd  = d + r / 3 - 1;
    int sh  = h + r % 3 - 1;
    int wg  = w0 - 1 + pi;
    bool valid = (unsigned)sd < DD && (unsigned)sh < HH && (unsigned)wg < WW;
#if USE_ASYNC
    if (valid) {
      size_t pos = ((size_t)((b * DD + sd) * HH + sh)) * WW + wg;
      async_g2l_b128(act + pos * 32 + q * 8, &sm[r][pi][q * 8]);
    } else {
      *(uint4*)(&sm[r][pi][q * 8]) = make_uint4(0, 0, 0, 0);
    }
#else
    uint4 v = make_uint4(0, 0, 0, 0);
    if (valid) {
      size_t pos = ((size_t)((b * DD + sd) * HH + sh)) * WW + wg;
      v = *(const uint4*)(act + pos * 32 + q * 8);
    }
    *(uint4*)(&sm[r][pi][q * 8]) = v;
#endif
  }
#if USE_ASYNC
  __builtin_amdgcn_s_wait_asynccnt(0);
#endif
  __syncthreads();

  int wave  = tid >> 5;            // wave32
  int lane  = tid & 31;
  int n     = lane & 15;           // output column within tile
  int csel  = (lane >> 4) << 4;    // K half: channels 0..15 or 16..31
  int wbase = wave << 4;

  v8f c0 = {}; v8f c1 = {};
  #pragma unroll
  for (int t = 0; t < 27; ++t) {
    int dz = t / 9, dy = (t / 3) % 3, dx = t % 3;
    int r = dz * 3 + dy;
    const v16bf bmat = *(const v16bf*)(&sm[r][wbase + n + dx][csel]);
    const v16bf a0 = *(const v16bf*)(wp + ((size_t)(0 * 27 + t) * 32 + lane) * 16);
    const v16bf a1 = *(const v16bf*)(wp + ((size_t)(1 * 27 + t) * 32 + lane) * 16);
    c0 = __builtin_amdgcn_wmma_f32_16x16x32_bf16(false, a0, false, bmat, (short)0, c0, false, false);
    c1 = __builtin_amdgcn_wmma_f32_16x16x32_bf16(false, a1, false, bmat, (short)0, c1, false, false);
  }

  // C/D layout: lane<16 -> rows M=r (ch 0..7 / 16..23); lane>=16 -> M=8+r.
  size_t pos = ((size_t)((b * DD + d) * HH + h)) * WW + (w0 + wbase + n);
  int chb = (lane < 16) ? 0 : 8;
  *(v8f*)(y + pos * 32 + chb)      = c0;
  *(v8f*)(y + pos * 32 + 16 + chb) = c1;

  // BN training stats: reduce across the 16 columns (shfl within half-wave),
  // then LDS float atomics, then one global float atomic per channel per block.
  v8f q0 = c0 * c0;
  v8f q1 = c1 * c1;
  #pragma unroll
  for (int off = 1; off < 16; off <<= 1) {
    #pragma unroll
    for (int rr = 0; rr < 8; ++rr) {
      c0[rr] += __shfl_xor(c0[rr], off, 32);
      c1[rr] += __shfl_xor(c1[rr], off, 32);
      q0[rr] += __shfl_xor(q0[rr], off, 32);
      q1[rr] += __shfl_xor(q1[rr], off, 32);
    }
  }
  if (n == 0) {
    #pragma unroll
    for (int rr = 0; rr < 8; ++rr) {
      atomicAdd(&sred[chb + rr],      c0[rr]);   // sum ch 0..15
      atomicAdd(&sred[16 + chb + rr], c1[rr]);   // sum ch 16..31
      atomicAdd(&sred[32 + chb + rr], q0[rr]);   // sumsq ch 0..15
      atomicAdd(&sred[48 + chb + rr], q1[rr]);   // sumsq ch 16..31
    }
  }
  __syncthreads();
  if (tid < 64) atomicAdd(&stats[tid], sred[tid]);
}

// ---------------------------------------------------------------------------
// BatchNorm (batch stats) + LeakyReLU(0.2), f32 -> bf16 channels-last repack.
__global__ void k_bnorm(const float* __restrict__ y, unsigned short* __restrict__ act,
                        const float* __restrict__ stats,
                        const float* __restrict__ gamma, const float* __restrict__ beta) {
  int id  = blockIdx.x * 256 + threadIdx.x;   // NPOS*4 chunks of 8 channels
  int pos = id >> 2;
  int cb  = (id & 3) * 8;
  const float inv_n = 1.0f / (float)NPOS;
  const v8f x = *(const v8f*)(y + (size_t)pos * 32 + cb);
  unsigned int pk[4];
  #pragma unroll
  for (int i = 0; i < 8; ++i) {
    int c = cb + i;
    float mean = stats[c] * inv_n;
    float var  = stats[32 + c] * inv_n - mean * mean;
    float sc   = gamma[c] * rsqrtf(var + 1e-5f);
    float sh   = beta[c] - mean * sc;
    float v    = x[i] * sc + sh;
    v = (v > 0.f) ? v : 0.2f * v;
    unsigned short hv = f2bf(v);
    if (i & 1) pk[i >> 1] |= ((unsigned int)hv) << 16; else pk[i >> 1] = hv;
  }
  *(uint4*)(act + (size_t)pos * 32 + cb) = make_uint4(pk[0], pk[1], pk[2], pk[3]);
}

// ---------------------------------------------------------------------------
// Final conv 32 -> 1 (+bias): dot of 864 taps per output; weights in LDS.
__global__ void k_final(const unsigned short* __restrict__ act,
                        const float* __restrict__ wf, const float* __restrict__ bfp,
                        float* __restrict__ out) {
  __shared__ float swf[32 * 27];
  int tid = threadIdx.x;
  for (int i = tid; i < 32 * 27; i += 256) swf[i] = wf[i];
  __syncthreads();
  int pos = blockIdx.x * 256 + tid;
  int w = pos & 127;
  int h = (pos >> 7) & 63;
  int t2 = pos >> 13;
  int d = t2 % DD;
  int b = t2 / DD;
  float acc = bfp[0];
  for (int t = 0; t < 27; ++t) {
    int sd = d + t / 9 - 1;
    int sh = h + (t / 3) % 3 - 1;
    int sw = w + t % 3 - 1;
    if ((unsigned)sd < DD && (unsigned)sh < HH && (unsigned)sw < WW) {
      const unsigned short* ap = act + (((size_t)((b * DD + sd) * HH + sh)) * WW + sw) * 32;
      uint4 u0 = ((const uint4*)ap)[0];
      uint4 u1 = ((const uint4*)ap)[1];
      uint4 u2 = ((const uint4*)ap)[2];
      uint4 u3 = ((const uint4*)ap)[3];
      unsigned int uu[16] = {u0.x,u0.y,u0.z,u0.w, u1.x,u1.y,u1.z,u1.w,
                             u2.x,u2.y,u2.z,u2.w, u3.x,u3.y,u3.z,u3.w};
      #pragma unroll
      for (int c = 0; c < 32; ++c) {
        unsigned short hv = (c & 1) ? (unsigned short)(uu[c >> 1] >> 16)
                                    : (unsigned short)(uu[c >> 1] & 0xFFFFu);
        acc += bf2f(hv) * swf[c * 27 + t];
      }
    }
  }
  out[pos] = acc;
}

// ---------------------------------------------------------------------------
extern "C" void kernel_launch(void* const* d_in, const int* in_sizes, int n_in,
                              void* d_out, int out_size, void* d_ws, size_t ws_size,
                              hipStream_t stream) {
  (void)in_sizes; (void)n_in; (void)out_size; (void)ws_size;
  const float* L   = (const float*)d_in[0];
  const float* R   = (const float*)d_in[1];
  const float* w1  = (const float*)d_in[2];
  const float* g1  = (const float*)d_in[3];
  const float* b1  = (const float*)d_in[4];
  const float* w2  = (const float*)d_in[5];
  const float* g2  = (const float*)d_in[6];
  const float* b2  = (const float*)d_in[7];
  const float* w3  = (const float*)d_in[8];
  const float* g3  = (const float*)d_in[9];
  const float* b3  = (const float*)d_in[10];
  const float* w4  = (const float*)d_in[11];
  const float* g4  = (const float*)d_in[12];
  const float* b4  = (const float*)d_in[13];
  const float* wf  = (const float*)d_in[14];
  const float* bfp = (const float*)d_in[15];

  // workspace layout
  char* ws = (char*)d_ws;
  unsigned short* ACT = (unsigned short*)ws;                          // 50,331,648 B
  float*          Y   = (float*)(ws + 50331648);                      // 100,663,296 B
  unsigned short* WP  = (unsigned short*)(ws + 50331648 + 100663296); // 221,184 B
  float*          ST  = (float*)(ws + 50331648 + 100663296 + 221184); // 1,024 B

  k_init  <<<1, 256, 0, stream>>>(ST);
  k_pack  <<<(4 * PACKSZ + 255) / 256, 256, 0, stream>>>(w1, w2, w3, w4, WP);
  k_volume<<<BB * HH * 2, 256, 0, stream>>>(L, R, ACT);

  const float* gs[4] = {g1, g2, g3, g4};
  const float* bs[4] = {b1, b2, b3, b4};
  for (int l = 0; l < 4; ++l) {
    k_conv <<<BB * DD * HH * 2, 128, 0, stream>>>(ACT, Y, WP + (size_t)l * PACKSZ, ST + l * 64);
    k_bnorm<<<NPOS * 4 / 256, 256, 0, stream>>>(Y, ACT, ST + l * 64, gs[l], bs[l]);
  }
  k_final<<<NPOS / 256, 256, 0, stream>>>(ACT, wf, bfp, (float*)d_out);
}
